// LSTMModel_68487548502232
// MI455X (gfx1250) — matrix operand order repaired and checked by reading, not compiled, not run on hardware
//
#include <hip/hip_runtime.h>
#include <hip/hip_bf16.h>
#include <math.h>

typedef _Float16 f16;
typedef __attribute__((ext_vector_type(16))) _Float16 v16h;
typedef __attribute__((ext_vector_type(8)))  _Float16 v8h;
typedef __attribute__((ext_vector_type(8)))  float    v8f;

#define B_  256
#define T_  512
#define F_  64
#define U_  128
#define G_  512   // 4*U
#define OUT_ 6
#define M_TOT (B_ * T_)   // 131072

__device__ inline v8f wmma_f16(v16h a, v16h b, v8f c) {
    // D = A(16x32 f16) * B(32x16 f16) + C(16x16 f32)
    return __builtin_amdgcn_wmma_f32_16x16x32_f16(
        false, a, false, b, (short)0, c, false, false);
}

__device__ inline v8f zero_v8f() {
    v8f z = {0.f, 0.f, 0.f, 0.f, 0.f, 0.f, 0.f, 0.f};
    return z;
}

__device__ inline v16h join16(v8h lo, v8h hi) {
    return __builtin_shufflevector(lo, hi, 0, 1, 2, 3, 4, 5, 6, 7,
                                           8, 9, 10, 11, 12, 13, 14, 15);
}

// Fast gate activations: one v_exp_f32 + v_rcp_f32 each.
__device__ inline float fast_sigmoid(float x) {
    return __builtin_amdgcn_rcpf(1.f + __expf(-x));
}
__device__ inline float fast_tanh(float x) {
    // tanh(x) = 1 - 2/(exp(2x)+1); exp saturates cleanly for the |z|<~30 range here
    return 1.f - 2.f * __builtin_amdgcn_rcpf(1.f + __expf(2.f * x));
}

// ---------------------------------------------------------------------------
// Swizzle f32 weight [K,512] into per-lane-contiguous WMMA B fragments (f16).
// dst element tid:  j = tid&15 (fragment elem), L = (tid>>4)&31 (lane),
//                   tile = tid>>9, nt = tile&31 (N tile), kk = tile>>5 (K tile)
// source (k,n):     k = kk*32 + (L>>4)*16 + j ; n = nt*16 + (L&15)
// ---------------------------------------------------------------------------
__global__ void swizzle_w_kernel(const float* __restrict__ src, f16* __restrict__ dst,
                                 int total) {
    const int tid = blockIdx.x * 256 + threadIdx.x;
    if (tid >= total) return;
    const int j    = tid & 15;
    const int L    = (tid >> 4) & 31;
    const int tile = tid >> 9;
    const int nt   = tile & 31;
    const int kk   = tile >> 5;
    const int k = kk * 32 + (L >> 4) * 16 + j;
    const int n = nt * 16 + (L & 15);
    dst[tid] = (f16)src[(size_t)k * G_ + n];
}

// ---------------------------------------------------------------------------
// XW = A[M,K] @ W[K,512] + bias   (WMMA; one wave computes a 16x64 tile)
// A is f32 (x) when AF16==false, f16 (H) when AF16==true; row stride == K.
// Wswz is the pre-swizzled fragment layout from swizzle_w_kernel.
// ---------------------------------------------------------------------------
template<int K, bool AF16>
__global__ __launch_bounds__(256)
void xw_gemm_kernel(const void* __restrict__ A_, const f16* __restrict__ Wswz,
                    const float* __restrict__ bias, float* __restrict__ XW) {
    const int lane  = threadIdx.x & 31;
    const int warp  = threadIdx.x >> 5;
    const int wave  = blockIdx.x * 8 + warp;
    const int m0    = (wave >> 3) * 16;   // 8192 row tiles
    const int n0    = (wave & 7) * 64;    // 8 col tiles of 64
    const int row   = lane & 15;
    const int hi    = lane >> 4;

    v8f acc[4];
#pragma unroll
    for (int t = 0; t < 4; ++t) acc[t] = zero_v8f();

#pragma unroll
    for (int kk = 0; kk < K / 32; ++kk) {
        const int k0 = kk * 32;
        v16h a;
        if (AF16) {
            const f16* ar = (const f16*)A_ + (size_t)(m0 + row) * K;
            v8h a0 = *(const v8h*)(ar + k0 + hi * 16);
            v8h a1 = *(const v8h*)(ar + k0 + hi * 16 + 8);
            a = join16(a0, a1);
        } else {
            const float* ar = (const float*)A_ + (size_t)(m0 + row) * K;
            v8f f0 = *(const v8f*)(ar + k0 + hi * 16);
            v8f f1 = *(const v8f*)(ar + k0 + hi * 16 + 8);
            a = join16(__builtin_convertvector(f0, v8h),
                       __builtin_convertvector(f1, v8h));
        }
#pragma unroll
        for (int t = 0; t < 4; ++t) {
            const v16h b = *(const v16h*)(
                Wswz + ((size_t)(kk * 32 + (n0 >> 4) + t) * 32 + lane) * 16);
            acc[t] = wmma_f16(a, b, acc[t]);
        }
    }

    const int rb = hi * 8;
#pragma unroll
    for (int t = 0; t < 4; ++t) {
        const int n = n0 + t * 16 + row;
        const float bn = bias[n];
#pragma unroll
        for (int r = 0; r < 8; ++r) {
            // XW is a write-once stream (268 MB): keep it out of L2
            __builtin_nontemporal_store(acc[t][r] + bn,
                                        &XW[(size_t)(m0 + rb + r) * G_ + n]);
        }
    }
}

// ---------------------------------------------------------------------------
// Persistent recurrent LSTM layer.
// Grid = 16 workgroups (16 batch rows each), 256 threads = 8 waves.
// Wave w owns gate columns [w*64, w*64+64) of z and holds R[:,w*64:w*64+64)
// in VGPRs as 16 pre-swizzled WMMA B fragments (loaded once, b128s).
// h ping-pongs through LDS (f16, A-fragment friendly), c lives in registers.
// ---------------------------------------------------------------------------
template<bool STORE_ALL>
__global__ __launch_bounds__(256, 1)
void lstm_recurrent_kernel(const float* __restrict__ XW, const f16* __restrict__ Rswz,
                           f16* __restrict__ Hout, float* __restrict__ hlast) {
    constexpr int ZS = 516;  // zbuf row stride in floats (bank-conflict padding)
    constexpr int HS = 136;  // hbuf row stride in f16   (bank-conflict padding)
    __shared__ float zbuf[16 * ZS];   // ~33 KB
    __shared__ f16   hbuf[16 * HS];   // ~4.3 KB

    const int lane = threadIdx.x & 31;
    const int w    = threadIdx.x >> 5;   // wave 0..7
    const int m0   = blockIdx.x * 16;    // batch tile
    const int row  = lane & 15;
    const int hi   = lane >> 4;
    const int n0   = w * 64;

    // ---- preload recurrent weight fragments (one-time, contiguous b128s) ----
    v16h Breg[4][4];
#pragma unroll
    for (int kk = 0; kk < 4; ++kk) {
#pragma unroll
        for (int t = 0; t < 4; ++t) {
            Breg[kk][t] = *(const v16h*)(
                Rswz + ((size_t)(kk * 32 + (n0 >> 4) + t) * 32 + lane) * 16);
        }
    }

    // ---- init state ----
    for (int i = threadIdx.x; i < 16 * HS; i += 256) hbuf[i] = (f16)0.f;
    float creg[8];
#pragma unroll
    for (int j = 0; j < 8; ++j) creg[j] = 0.f;
    __syncthreads();

    for (int tt = 0; tt < T_; ++tt) {
        // ---- fetch this step's input projection (overlaps with WMMA below) ----
        float xwv[4][8];
#pragma unroll
        for (int t = 0; t < 4; ++t) {
            const int n = n0 + t * 16 + row;
#pragma unroll
            for (int r = 0; r < 8; ++r) {
                const int b = m0 + hi * 8 + r;
                xwv[t][r] = __builtin_nontemporal_load(
                    &XW[((size_t)b * T_ + tt) * G_ + n]);
            }
        }
        if (tt + 1 < T_) {
            // prefetch next step's slab (global_prefetch_b8)
            const int b = m0 + hi * 8;
            __builtin_prefetch(&XW[((size_t)b * T_ + (tt + 1)) * G_ + n0 + row], 0, 1);
        }

        // ---- z = h @ R (WMMA) ----
        v8f acc[4];
#pragma unroll
        for (int t = 0; t < 4; ++t) acc[t] = zero_v8f();
#pragma unroll
        for (int kk = 0; kk < 4; ++kk) {
            const int k0 = kk * 32;
            const f16* hb = &hbuf[row * HS];
            v8h a0 = *(const v8h*)(hb + k0 + hi * 16);      // ds_load_b128
            v8h a1 = *(const v8h*)(hb + k0 + hi * 16 + 8);  // ds_load_b128
            const v16h a = join16(a0, a1);
#pragma unroll
            for (int t = 0; t < 4; ++t)
                acc[t] = wmma_f16(a, Breg[kk][t], acc[t]);
        }

        // ---- z += xw, publish to LDS ----
#pragma unroll
        for (int t = 0; t < 4; ++t) {
            const int n = n0 + t * 16 + row;
#pragma unroll
            for (int r = 0; r < 8; ++r)
                zbuf[(hi * 8 + r) * ZS + n] = acc[t][r] + xwv[t][r];
        }
        __syncthreads();

        // ---- gates: wave w owns unit columns [w*16, w*16+16) ----
#pragma unroll
        for (int j = 0; j < 8; ++j) {
            const int u  = w * 16 + j * 2 + hi;
            const float zi = zbuf[row * ZS + u];
            const float zf = zbuf[row * ZS + U_ + u];
            const float zg = zbuf[row * ZS + 2 * U_ + u];
            const float zo = zbuf[row * ZS + 3 * U_ + u];
            const float ig = fast_sigmoid(zi);
            const float fg = fast_sigmoid(zf);
            const float gg = fast_tanh(zg);
            const float og = fast_sigmoid(zo);
            const float cv = fg * creg[j] + ig * gg;
            creg[j] = cv;
            const float hv = og * fast_tanh(cv);
            hbuf[row * HS + u] = (f16)hv;
            if (STORE_ALL) {
                Hout[((size_t)(m0 + row) * T_ + tt) * U_ + u] = (f16)hv;
            } else if (tt == T_ - 1) {
                hlast[(m0 + row) * U_ + u] = hv;
            }
        }
        __syncthreads();
    }
}

// ---------------------------------------------------------------------------
// Dense head: out[256,6] = hlast[256,128] @ Wd[128,6] + bd
// ---------------------------------------------------------------------------
__global__ void dense_head_kernel(const float* __restrict__ hl, const float* __restrict__ Wd,
                                  const float* __restrict__ bd, float* __restrict__ out) {
    const int b = blockIdx.x * blockDim.x + threadIdx.x;
    if (b >= B_) return;
    float acc[OUT_];
#pragma unroll
    for (int o = 0; o < OUT_; ++o) acc[o] = bd[o];
    for (int k = 0; k < U_; ++k) {
        const float h = hl[b * U_ + k];
#pragma unroll
        for (int o = 0; o < OUT_; ++o) acc[o] += h * Wd[k * OUT_ + o];
    }
#pragma unroll
    for (int o = 0; o < OUT_; ++o) out[b * OUT_ + o] = acc[o];
}

// ---------------------------------------------------------------------------
extern "C" void kernel_launch(void* const* d_in, const int* in_sizes, int n_in,
                              void* d_out, int out_size, void* d_ws, size_t ws_size,
                              hipStream_t stream) {
    const float* x  = (const float*)d_in[0];
    const float* W0 = (const float*)d_in[1];
    const float* R0 = (const float*)d_in[2];
    const float* b0 = (const float*)d_in[3];
    const float* W1 = (const float*)d_in[4];
    const float* R1 = (const float*)d_in[5];
    const float* b1 = (const float*)d_in[6];
    const float* W2 = (const float*)d_in[7];
    const float* R2 = (const float*)d_in[8];
    const float* b2 = (const float*)d_in[9];
    const float* Wd = (const float*)d_in[10];
    const float* bd = (const float*)d_in[11];
    float* out = (float*)d_out;

    char* ws = (char*)d_ws;
    size_t off = 0;
    float* XW = (float*)(ws + off);  off += (size_t)M_TOT * G_ * sizeof(float);   // 268 MB
    f16*   H  = (f16*)(ws + off);    off += (size_t)M_TOT * U_ * sizeof(f16);     // 33.5 MB
    float* hl = (float*)(ws + off);  off += (size_t)B_ * U_ * sizeof(float);
    f16* W0s = (f16*)(ws + off);     off += (size_t)F_ * G_ * sizeof(f16);
    f16* R0s = (f16*)(ws + off);     off += (size_t)U_ * G_ * sizeof(f16);
    f16* W1s = (f16*)(ws + off);     off += (size_t)U_ * G_ * sizeof(f16);
    f16* R1s = (f16*)(ws + off);     off += (size_t)U_ * G_ * sizeof(f16);
    f16* W2s = (f16*)(ws + off);     off += (size_t)U_ * G_ * sizeof(f16);
    f16* R2s = (f16*)(ws + off);     off += (size_t)U_ * G_ * sizeof(f16);

    // weights -> pre-swizzled f16 WMMA B fragments
    swizzle_w_kernel<<<(F_ * G_ + 255) / 256, 256, 0, stream>>>(W0, W0s, F_ * G_);
    swizzle_w_kernel<<<(U_ * G_ + 255) / 256, 256, 0, stream>>>(R0, R0s, U_ * G_);
    swizzle_w_kernel<<<(U_ * G_ + 255) / 256, 256, 0, stream>>>(W1, W1s, U_ * G_);
    swizzle_w_kernel<<<(U_ * G_ + 255) / 256, 256, 0, stream>>>(R1, R1s, U_ * G_);
    swizzle_w_kernel<<<(U_ * G_ + 255) / 256, 256, 0, stream>>>(W2, W2s, U_ * G_);
    swizzle_w_kernel<<<(U_ * G_ + 255) / 256, 256, 0, stream>>>(R2, R2s, U_ * G_);

    const int gemm_blocks = 8192;  // 8 waves each; covers 131072 rows x 512 cols

    // layer 0
    xw_gemm_kernel<F_, false><<<gemm_blocks, 256, 0, stream>>>(x, W0s, b0, XW);
    lstm_recurrent_kernel<true><<<B_ / 16, 256, 0, stream>>>(XW, R0s, H, nullptr);
    // layer 1
    xw_gemm_kernel<U_, true><<<gemm_blocks, 256, 0, stream>>>(H, W1s, b1, XW);
    lstm_recurrent_kernel<true><<<B_ / 16, 256, 0, stream>>>(XW, R1s, H, nullptr);
    // layer 2 (last h only)
    xw_gemm_kernel<U_, true><<<gemm_blocks, 256, 0, stream>>>(H, W2s, b2, XW);
    lstm_recurrent_kernel<false><<<B_ / 16, 256, 0, stream>>>(XW, R2s, nullptr, hl);
    // dense head
    dense_head_kernel<<<1, 256, 0, stream>>>(hl, Wd, bd, out);
}
